// TransformerBlock_12764642803911
// MI455X (gfx1250) — matrix operand order repaired and checked by reading, not compiled, not run on hardware
//
#include <hip/hip_runtime.h>

// ---------------------------------------------------------------------------
// GPT-2 transformer block for MI455X (gfx1250), bf16 WMMA + fp32 accumulate.
// Async global->LDS staging (ASYNCcnt), double buffering, fused QKV GEMM.
// ---------------------------------------------------------------------------

typedef __attribute__((ext_vector_type(16))) __bf16        v16bf;
typedef __attribute__((ext_vector_type(8)))  float          v8f;
typedef __attribute__((ext_vector_type(8)))  unsigned short us8;
typedef __attribute__((ext_vector_type(16))) unsigned short us16;

union BF16Frag {
    us16  u;
    us8   h[2];
    v16bf bf;
};

// ---- CDNA5 async copy support (compile-time feature detect) ---------------
#if defined(__has_builtin)
#  if __has_builtin(__builtin_amdgcn_global_load_async_to_lds_b128)
#    define HAVE_ASYNC_COPY 1
#  else
#    define HAVE_ASYNC_COPY 0
#  endif
#  if __has_builtin(__builtin_amdgcn_s_wait_asynccnt)
#    define HAVE_WAIT_ASYNC 1
#  else
#    define HAVE_WAIT_ASYNC 0
#  endif
#else
#  define HAVE_ASYNC_COPY 0
#  define HAVE_WAIT_ASYNC 0
#endif

// Builtin signature (from clang): (int4 addrspace(1)*, int4 addrspace(3)*, Ii, Ii)
typedef int i32v4 __attribute__((vector_size(4 * sizeof(int))));
#define GPTR(p) ((__attribute__((address_space(1))) i32v4*)(p))
#define LPTR(p) ((__attribute__((address_space(3))) i32v4*)(p))

__device__ __forceinline__ void wait_async0() {
#if HAVE_WAIT_ASYNC
    __builtin_amdgcn_s_wait_asynccnt(0);
#else
    asm volatile("s_wait_asynccnt 0x0" ::: "memory");
#endif
}

__device__ __forceinline__ unsigned short f32_to_bf16(float f) {
    unsigned int u = __float_as_uint(f);
    u += 0x7FFFu + ((u >> 16) & 1u);      // round-to-nearest-even
    return (unsigned short)(u >> 16);
}

__device__ __forceinline__ float gelu_tanh(float x) {
    const float c = 0.7978845608028654f;  // sqrt(2/pi)
    float x3 = x * x * x;
    return 0.5f * x * (1.0f + tanhf(c * (x + 0.044715f * x3)));
}

// ---------------------------------------------------------------------------
// fp32 -> bf16 elementwise convert (weights)
// ---------------------------------------------------------------------------
__global__ void convert_f32_bf16(const float* __restrict__ in,
                                 unsigned short* __restrict__ out, int n) {
    int i = blockIdx.x * blockDim.x + threadIdx.x;
    if (i < n) out[i] = f32_to_bf16(in[i]);
}

// ---------------------------------------------------------------------------
// LayerNorm over 768 columns, one block (256 threads = 8 waves) per row.
// ---------------------------------------------------------------------------
__global__ __launch_bounds__(256)
void layernorm_bf16(const float* __restrict__ x,
                    const float* __restrict__ sc,
                    const float* __restrict__ sh,
                    unsigned short* __restrict__ out) {
    __shared__ float wsum[8];
    __shared__ float wvar[8];
    const int row  = blockIdx.x;
    const int tid  = threadIdx.x;
    const int wave = __builtin_amdgcn_readfirstlane(tid >> 5);
    const int lane = tid & 31;
    const float* xr = x + (size_t)row * 768;

    float vals[3];
    float s = 0.f;
#pragma unroll
    for (int i = 0; i < 3; ++i) { vals[i] = xr[tid + i * 256]; s += vals[i]; }
#pragma unroll
    for (int off = 1; off < 32; off <<= 1) s += __shfl_xor(s, off, 32);
    if (lane == 0) wsum[wave] = s;
    __syncthreads();
    float tot = 0.f;
#pragma unroll
    for (int w = 0; w < 8; ++w) tot += wsum[w];
    const float mean = tot * (1.0f / 768.0f);

    float vs = 0.f;
#pragma unroll
    for (int i = 0; i < 3; ++i) { float d = vals[i] - mean; vs += d * d; }
#pragma unroll
    for (int off = 1; off < 32; off <<= 1) vs += __shfl_xor(vs, off, 32);
    if (lane == 0) wvar[wave] = vs;
    __syncthreads();
    float vtot = 0.f;
#pragma unroll
    for (int w = 0; w < 8; ++w) vtot += wvar[w];
    const float rstd = rsqrtf(vtot * (1.0f / 768.0f) + 1e-5f);

#pragma unroll
    for (int i = 0; i < 3; ++i) {
        int c = tid + i * 256;
        out[(size_t)row * 768 + c] =
            f32_to_bf16((vals[i] - mean) * rstd * sc[c] + sh[c]);
    }
}

// ---------------------------------------------------------------------------
// NT GEMM: C[M,N] = A[M,K] * W[N,K]^T  (bf16 in, fp32 accumulate)
// Block tile 128x256, BK=32, 8 waves (2x4), wave tile 64x64 (4x4 WMMA accs).
// Double-buffered LDS; async global->LDS tile staging when available.
// EPI: 0 = bf16 store, 1 = fp32 store + residual add, 2 = GELU -> bf16 store
// ---------------------------------------------------------------------------
template <int EPI>
__global__ __launch_bounds__(256)
void gemm_nt_bf16(const unsigned short* __restrict__ A,
                  const unsigned short* __restrict__ W,
                  void* __restrict__ outv,
                  const float* __restrict__ res,
                  int M, int N, int K) {
    constexpr int BM = 128, BN = 256, BK = 32, LDP = BK + 8; // 80B row stride
    __shared__ __align__(16) unsigned short As[2][BM][LDP];
    __shared__ __align__(16) unsigned short Bs[2][BN][LDP];

    const int tid   = threadIdx.x;
    const int wave  = __builtin_amdgcn_readfirstlane(tid >> 5);
    const int lane  = tid & 31;
    const int lhalf = lane >> 4;
    const int l16   = lane & 15;

    const int bm0 = blockIdx.y * BM;
    const int bn0 = blockIdx.x * BN;
    const int wm  = wave >> 2;   // 0..1 -> 64 rows
    const int wn  = wave & 3;    // 0..3 -> 64 cols

    // global->LDS staging: one b128 = 8 bf16 (16B) per slot
    const int ra = tid >> 2;        // 0..63
    const int c8 = (tid & 3) * 8;   // 0,8,16,24
    const unsigned short* Abase = A + (size_t)bm0 * K + c8;
    const unsigned short* Wbase = W + (size_t)bn0 * K + c8;

#if HAVE_ASYNC_COPY
    auto stage_tile = [&](int k0, int buf) {
#pragma unroll
        for (int i = 0; i < 2; ++i)
            __builtin_amdgcn_global_load_async_to_lds_b128(
                GPTR(Abase + (size_t)(ra + i * 64) * K + k0),
                LPTR(&As[buf][ra + i * 64][c8]), 0, 0);
#pragma unroll
        for (int i = 0; i < 4; ++i)
            __builtin_amdgcn_global_load_async_to_lds_b128(
                GPTR(Wbase + (size_t)(ra + i * 64) * K + k0),
                LPTR(&Bs[buf][ra + i * 64][c8]), 0, 0);
    };
#else
    uint4 areg[2], breg[4];
    auto load_regs = [&](int k0) {
#pragma unroll
        for (int i = 0; i < 2; ++i)
            areg[i] = *(const uint4*)(Abase + (size_t)(ra + i * 64) * K + k0);
#pragma unroll
        for (int i = 0; i < 4; ++i)
            breg[i] = *(const uint4*)(Wbase + (size_t)(ra + i * 64) * K + k0);
    };
    auto store_lds = [&](int buf) {
#pragma unroll
        for (int i = 0; i < 2; ++i)
            *(uint4*)&As[buf][ra + i * 64][c8] = areg[i];
#pragma unroll
        for (int i = 0; i < 4; ++i)
            *(uint4*)&Bs[buf][ra + i * 64][c8] = breg[i];
    };
#endif

    v8f acc[4][4];
#pragma unroll
    for (int mt = 0; mt < 4; ++mt)
#pragma unroll
        for (int nt = 0; nt < 4; ++nt)
#pragma unroll
            for (int i = 0; i < 8; ++i) acc[mt][nt][i] = 0.0f;

#if HAVE_ASYNC_COPY
    stage_tile(0, 0);
    wait_async0();
#else
    load_regs(0);
    store_lds(0);
#endif
    __syncthreads();

    const int nk = K / BK;
    for (int kk = 0; kk < nk; ++kk) {
        const int cur = kk & 1;
        if (kk + 1 < nk) {
#if HAVE_ASYNC_COPY
            stage_tile((kk + 1) * BK, cur ^ 1);
#else
            load_regs((kk + 1) * BK);
#endif
            if (kk + 2 < nk) {
                __builtin_prefetch(Abase + (size_t)ra * K + (kk + 2) * BK, 0, 1);
                __builtin_prefetch(Wbase + (size_t)ra * K + (kk + 2) * BK, 0, 1);
            }
        }

        BF16Frag afr[4], bfr[4];
        const int kbA = lhalf * 16;   // A layout: K kbA..kbA+15 contiguous
        const int kbB = lhalf * 16;   // B layout: K kbB..kbB+15 contiguous
#pragma unroll
        for (int mt = 0; mt < 4; ++mt) {
            int r = wm * 64 + mt * 16 + l16;
            afr[mt].h[0] = *(const us8*)&As[cur][r][kbA];
            afr[mt].h[1] = *(const us8*)&As[cur][r][kbA + 8];
        }
#pragma unroll
        for (int nt = 0; nt < 4; ++nt) {
            int r = wn * 64 + nt * 16 + l16;
            bfr[nt].h[0] = *(const us8*)&Bs[cur][r][kbB];
            bfr[nt].h[1] = *(const us8*)&Bs[cur][r][kbB + 8];
        }
#pragma unroll
        for (int mt = 0; mt < 4; ++mt)
#pragma unroll
            for (int nt = 0; nt < 4; ++nt)
                acc[mt][nt] = __builtin_amdgcn_wmma_f32_16x16x32_bf16(
                    false, afr[mt].bf, false, bfr[nt].bf,
                    (short)0, acc[mt][nt], false, false);

        if (kk + 1 < nk) {
#if HAVE_ASYNC_COPY
            wait_async0();
#else
            store_lds(cur ^ 1);
#endif
        }
        __syncthreads();
    }

    // Epilogue. C layout: VGPR i -> (rowtile + i + 8*lhalf, coltile + l16)
#pragma unroll
    for (int mt = 0; mt < 4; ++mt)
#pragma unroll
        for (int nt = 0; nt < 4; ++nt)
#pragma unroll
            for (int i = 0; i < 8; ++i) {
                int row = bm0 + wm * 64 + mt * 16 + i + 8 * lhalf;
                int col = bn0 + wn * 64 + nt * 16 + l16;
                size_t idx = (size_t)row * N + col;
                float v = acc[mt][nt][i];
                if (EPI == 0) {
                    ((unsigned short*)outv)[idx] = f32_to_bf16(v);
                } else if (EPI == 1) {
                    ((float*)outv)[idx] = v + res[idx];
                } else {
                    ((unsigned short*)outv)[idx] = f32_to_bf16(gelu_tanh(v));
                }
            }
}

// ---------------------------------------------------------------------------
// Causal flash attention over fused QKV activations.
// qkv: bf16 [B*T, 2304], Q at col 0, K at 768, V at 1536; head h at +h*64.
// K tiles async-copied to LDS; V transposed through registers.
// Softmax in exp2 domain; masking only on diagonal blocks (scalar branch).
// ---------------------------------------------------------------------------
__global__ __launch_bounds__(256)
void attn_flash(const unsigned short* __restrict__ qkv,
                unsigned short* __restrict__ O) {
    constexpr int T = 1024, E = 768, H = 12, D = 64, S = 3 * E; // row stride
    __shared__ __align__(16) unsigned short Ks[2][32][D + 8];     // 32 x 72
    __shared__ __align__(16) unsigned short Vts[2][D][32 + 8];    // V^T 64 x 40
    __shared__ __align__(16) unsigned short Ps[8][16][32 + 8];    // per-wave P

    const int bid = blockIdx.x;
    const int qb  = bid & 7;
    const int h   = (bid >> 3) % H;
    const int b   = bid / (8 * H);
    const int q0  = qb * 128;

    const int tid   = threadIdx.x;
    const int wave  = __builtin_amdgcn_readfirstlane(tid >> 5); // uniform
    const int lane  = tid & 31;
    const int lhalf = lane >> 4;
    const int l16   = lane & 15;
    const int qrow  = q0 + wave * 16;      // scalar per wave

    const unsigned short* Qp = qkv + h * D;
    const unsigned short* Kp = qkv + E + h * D;
    const unsigned short* Vp = qkv + 2 * E + h * D;

    // Q fragments (d 0..31 and 32..63), kept in registers for the whole loop.
    BF16Frag qf[2];
    {
        const size_t roff = (size_t)(b * T + qrow + l16) * S;
        const int kbA = lhalf * 16;
        qf[0].h[0] = *(const us8*)(Qp + roff + kbA);
        qf[0].h[1] = *(const us8*)(Qp + roff + kbA + 8);
        qf[1].h[0] = *(const us8*)(Qp + roff + 32 + kbA);
        qf[1].h[1] = *(const us8*)(Qp + roff + 32 + kbA + 8);
    }

    float m_run[8], l_run[8];
    v8f ctx[4];
#pragma unroll
    for (int i = 0; i < 8; ++i) { m_run[i] = -3.0e38f; l_run[i] = 0.0f; }
#pragma unroll
    for (int t = 0; t < 4; ++t)
#pragma unroll
        for (int i = 0; i < 8; ++i) ctx[t][i] = 0.0f;

    // scores scaled straight into log2 domain: 1/sqrt(64) * log2(e)
    const float sc2 = 0.125f * 1.4426950408889634f;
    const int nblk = (q0 + 128) / 32;

    const int c4 = (tid & 15) * 4;        // 0..60 (V staging)
    const int r  = tid >> 4;              // 0..15

#if HAVE_ASYNC_COPY
    const int kr8 = tid >> 3;             // 0..31 (K staging row)
    const int kc8 = (tid & 7) * 8;        // 0..56 (K staging col)
    auto stage_k = [&](int k0, int buf) {
        __builtin_amdgcn_global_load_async_to_lds_b128(
            GPTR(Kp + (size_t)(b * T + k0 + kr8) * S + kc8),
            LPTR(&Ks[buf][kr8][kc8]), 0, 0);
    };
#endif
    uint2 kreg[2], vreg[2];
    auto load_regs = [&](int k0) {
#pragma unroll
        for (int i = 0; i < 2; ++i) {
            size_t goff = (size_t)(b * T + k0 + i * 16 + r) * S + c4;
#if !HAVE_ASYNC_COPY
            kreg[i] = *(const uint2*)(Kp + goff);
#endif
            vreg[i] = *(const uint2*)(Vp + goff);
        }
    };
    auto store_lds = [&](int buf) {
#pragma unroll
        for (int i = 0; i < 2; ++i) {
            int rr = i * 16 + r;
#if !HAVE_ASYNC_COPY
            *(uint2*)&Ks[buf][rr][c4] = kreg[i];
#endif
            const unsigned short* vs = (const unsigned short*)&vreg[i];
#pragma unroll
            for (int e = 0; e < 4; ++e) Vts[buf][c4 + e][rr] = vs[e];
        }
    };

#if HAVE_ASYNC_COPY
    stage_k(0, 0);
#endif
    load_regs(0);
    store_lds(0);
#if HAVE_ASYNC_COPY
    wait_async0();
#endif
    __syncthreads();

    for (int j = 0; j < nblk; ++j) {
        const int k0  = j * 32;
        const int cur = j & 1;
        if (j + 1 < nblk) {
#if HAVE_ASYNC_COPY
            stage_k(k0 + 32, cur ^ 1);
#endif
            load_regs(k0 + 32);
            if (j + 2 < nblk) {
                __builtin_prefetch(Kp + (size_t)(b * T + k0 + 64 + r) * S + c4, 0, 1);
                __builtin_prefetch(Vp + (size_t)(b * T + k0 + 64 + r) * S + c4, 0, 1);
            }
        }

        if (k0 <= qrow + 15) {          // scalar (qrow uniform per wave)
            // S = Q K^T : two 16x16 score tiles over this 32-key block
            v8f s[2];
#pragma unroll
            for (int nt = 0; nt < 2; ++nt)
#pragma unroll
                for (int i = 0; i < 8; ++i) s[nt][i] = 0.0f;

#pragma unroll
            for (int nt = 0; nt < 2; ++nt) {
                BF16Frag kf0, kf1;
                int kr = nt * 16 + l16;           // score col = K row
                kf0.h[0] = *(const us8*)&Ks[cur][kr][lhalf * 16];
                kf0.h[1] = *(const us8*)&Ks[cur][kr][lhalf * 16 + 8];
                kf1.h[0] = *(const us8*)&Ks[cur][kr][32 + lhalf * 16];
                kf1.h[1] = *(const us8*)&Ks[cur][kr][32 + lhalf * 16 + 8];
                s[nt] = __builtin_amdgcn_wmma_f32_16x16x32_bf16(
                    false, qf[0].bf, false, kf0.bf, (short)0, s[nt], false, false);
                s[nt] = __builtin_amdgcn_wmma_f32_16x16x32_bf16(
                    false, qf[1].bf, false, kf1.bf, (short)0, s[nt], false, false);
            }

            // scale into log2 domain; mask only on diagonal blocks
            float m_blk[8];
            if (k0 + 31 > qrow) {       // scalar branch
#pragma unroll
                for (int i = 0; i < 8; ++i) {
                    int q = qrow + i + 8 * lhalf;
#pragma unroll
                    for (int nt = 0; nt < 2; ++nt) {
                        int kc = k0 + nt * 16 + l16;
                        float v = s[nt][i] * sc2;
                        if (kc > q) v = -3.0e38f;
                        s[nt][i] = v;
                    }
                    m_blk[i] = fmaxf(s[0][i], s[1][i]);
                }
            } else {
#pragma unroll
                for (int i = 0; i < 8; ++i) {
                    s[0][i] *= sc2;
                    s[1][i] *= sc2;
                    m_blk[i] = fmaxf(s[0][i], s[1][i]);
                }
            }
#pragma unroll
            for (int off = 1; off < 16; off <<= 1)
#pragma unroll
                for (int i = 0; i < 8; ++i)
                    m_blk[i] = fmaxf(m_blk[i], __shfl_xor(m_blk[i], off, 32));

            float alpha[8], l_blk[8];
#pragma unroll
            for (int i = 0; i < 8; ++i) {
                float mn = fmaxf(m_run[i], m_blk[i]);
                alpha[i] = exp2f(m_run[i] - mn);
                m_run[i] = mn;
                float p0 = exp2f(s[0][i] - mn);
                float p1 = exp2f(s[1][i] - mn);
                s[0][i] = p0; s[1][i] = p1;
                l_blk[i] = p0 + p1;
            }
#pragma unroll
            for (int off = 1; off < 16; off <<= 1)
#pragma unroll
                for (int i = 0; i < 8; ++i)
                    l_blk[i] += __shfl_xor(l_blk[i], off, 32);
#pragma unroll
            for (int i = 0; i < 8; ++i)
                l_run[i] = l_run[i] * alpha[i] + l_blk[i];
#pragma unroll
            for (int t = 0; t < 4; ++t)
#pragma unroll
                for (int i = 0; i < 8; ++i) ctx[t][i] *= alpha[i];

            // Restage P (C layout) -> per-wave LDS -> A-layout fragment
#pragma unroll
            for (int nt = 0; nt < 2; ++nt)
#pragma unroll
                for (int i = 0; i < 8; ++i)
                    Ps[wave][i + 8 * lhalf][nt * 16 + l16] = f32_to_bf16(s[nt][i]);

            BF16Frag pf;
            const int kbA = lhalf * 16;
            pf.h[0] = *(const us8*)&Ps[wave][l16][kbA];
            pf.h[1] = *(const us8*)&Ps[wave][l16][kbA + 8];

            // ctx += P @ V  (four 16-wide d tiles)
#pragma unroll
            for (int t = 0; t < 4; ++t) {
                BF16Frag vf;
                int d = t * 16 + l16;
                vf.h[0] = *(const us8*)&Vts[cur][d][lhalf * 16];
                vf.h[1] = *(const us8*)&Vts[cur][d][lhalf * 16 + 8];
                ctx[t] = __builtin_amdgcn_wmma_f32_16x16x32_bf16(
                    false, pf.bf, false, vf.bf, (short)0, ctx[t], false, false);
            }
        }

        if (j + 1 < nblk) {
            store_lds(cur ^ 1);
#if HAVE_ASYNC_COPY
            wait_async0();
#endif
        }
        __syncthreads();
    }

    // Normalize and store bf16 context ([M,768] layout for the proj GEMM)
#pragma unroll
    for (int i = 0; i < 8; ++i) {
        float inv = 1.0f / l_run[i];
#pragma unroll
        for (int t = 0; t < 4; ++t) {
            int row = qrow + i + 8 * lhalf;
            float v = ctx[t][i] * inv;
            O[(size_t)(b * T + row) * E + h * D + t * 16 + l16] = f32_to_bf16(v);
        }
    }
}

// ---------------------------------------------------------------------------
// Host orchestration
// ---------------------------------------------------------------------------
extern "C" void kernel_launch(void* const* d_in, const int* in_sizes, int n_in,
                              void* d_out, int out_size, void* d_ws, size_t ws_size,
                              hipStream_t stream) {
    (void)in_sizes; (void)n_in; (void)out_size; (void)ws_size;
    constexpr int Bm = 8, T = 1024, E = 768, F = 3072, QKV = 3 * E;
    constexpr int M = Bm * T;                 // 8192 rows
    const float* x         = (const float*)d_in[0];
    const float* wq        = (const float*)d_in[1];
    const float* wk        = (const float*)d_in[2];
    const float* wv        = (const float*)d_in[3];
    const float* wo        = (const float*)d_in[4];
    const float* w1        = (const float*)d_in[5];
    const float* w2        = (const float*)d_in[6];
    const float* ln1_scale = (const float*)d_in[7];
    const float* ln1_shift = (const float*)d_in[8];
    const float* ln2_scale = (const float*)d_in[9];
    const float* ln2_shift = (const float*)d_in[10];
    float* out = (float*)d_out;

    // Workspace carve-up (256B aligned)
    size_t off = 0;
    auto alloc_us = [&](size_t n) {
        unsigned short* p = (unsigned short*)((char*)d_ws + off);
        off += ((n * 2 + 255) / 256) * 256;
        return p;
    };
    auto alloc_f = [&](size_t n) {
        float* p = (float*)((char*)d_ws + off);
        off += ((n * 4 + 255) / 256) * 256;
        return p;
    };
    unsigned short* wqkv16 = alloc_us((size_t)QKV * E);   // [2304, 768]
    unsigned short* wo16   = alloc_us((size_t)E * E);
    unsigned short* w1_16  = alloc_us((size_t)F * E);
    unsigned short* w2_16  = alloc_us((size_t)E * F);
    unsigned short* h16    = alloc_us((size_t)M * E);
    unsigned short* qkv16  = alloc_us((size_t)M * QKV);   // [M, 2304]
    unsigned short* ctx16  = alloc_us((size_t)M * E);
    float*          x1     = alloc_f((size_t)M * E);
    unsigned short* act16  = alloc_us((size_t)M * F);

    // 1) weights -> bf16 (q/k/v concatenated along output dim)
    auto cvt = [&](const float* src, unsigned short* dst, int n) {
        convert_f32_bf16<<<(n + 255) / 256, 256, 0, stream>>>(src, dst, n);
    };
    cvt(wq, wqkv16, E * E);
    cvt(wk, wqkv16 + (size_t)E * E, E * E);
    cvt(wv, wqkv16 + (size_t)2 * E * E, E * E);
    cvt(wo, wo16, E * E);
    cvt(w1, w1_16, F * E);
    cvt(w2, w2_16, E * F);

    // 2) LN1
    layernorm_bf16<<<M, 256, 0, stream>>>(x, ln1_scale, ln1_shift, h16);

    // 3) fused QKV projection: [M,768] x [2304,768]^T -> [M,2304]
    dim3 gQKV(QKV / 256, M / 128);            // 9 x 64
    gemm_nt_bf16<0><<<gQKV, 256, 0, stream>>>(h16, wqkv16, qkv16, nullptr,
                                              M, QKV, E);

    // 4) causal attention
    attn_flash<<<Bm * 12 * (T / 128), 256, 0, stream>>>(qkv16, ctx16);

    // 5) output projection + residual -> x1 (fp32)
    dim3 gE(E / 256, M / 128);                // 3 x 64
    gemm_nt_bf16<1><<<gE, 256, 0, stream>>>(ctx16, wo16, x1, x, M, E, E);

    // 6) LN2
    layernorm_bf16<<<M, 256, 0, stream>>>(x1, ln2_scale, ln2_shift, h16);

    // 7) FFN up + GELU -> bf16
    dim3 gF(F / 256, M / 128);                // 12 x 64
    gemm_nt_bf16<2><<<gF, 256, 0, stream>>>(h16, w1_16, act16, nullptr, M, F, E);

    // 8) FFN down + residual -> fp32 out
    gemm_nt_bf16<1><<<gE, 256, 0, stream>>>(act16, w2_16, out, x1, M, E, F);
}
